// MultiHeadAttention_77489799954930
// MI455X (gfx1250) — matrix operand order repaired and checked by reading, not compiled, not run on hardware
//
#include <hip/hip_runtime.h>
#include <hip/hip_bf16.h>

// ---------------------------------------------------------------------------
// MHA forward for B=2, T=2048, D=2048, H=16, HD=128 on gfx1250 (MI455X).
// bf16 WMMA everywhere (v_wmma_f32_16x16x32_bf16), flash attention,
// async global->LDS double-buffered GEMM B tiles (ASYNCcnt path).
// ---------------------------------------------------------------------------

typedef __bf16 bf16;
typedef __attribute__((ext_vector_type(8)))  __bf16 v8bf;
typedef __attribute__((ext_vector_type(16))) __bf16 v16bf;
typedef __attribute__((ext_vector_type(8)))  float  v8f;

// exact pointee type expected by the async-LDS builtins (per hipcc diagnostic)
typedef int v4i_async __attribute__((__vector_size__(4 * sizeof(int))));
typedef __attribute__((address_space(1))) v4i_async* gptr_b128;
typedef __attribute__((address_space(3))) v4i_async* lptr_b128;

static constexpr int BSZ  = 2;
static constexpr int T    = 2048;
static constexpr int DM   = 2048;
static constexpr int H    = 16;
static constexpr int HD   = 128;
static constexpr int HALF = 64;
static constexpr int LDSP = 40;   // LDS pitch (bf16) for 32-wide P tiles
static constexpr int KSTEP = 64;  // GEMM k per LDS stage (2 WMMA k-chunks)
static constexpr int BP   = 72;   // LDS pitch (bf16) for 64-wide B tile rows

#if defined(__has_builtin)
#  if __has_builtin(__builtin_amdgcn_global_load_async_to_lds_b128) && \
      __has_builtin(__builtin_amdgcn_s_wait_asynccnt)
#    define USE_ASYNC_LDS 1
#  endif
#endif
#ifndef USE_ASYNC_LDS
#  define USE_ASYNC_LDS 0
#endif

__device__ inline void async_copy_b128(const bf16* gsrc, __bf16* ldst) {
#if USE_ASYNC_LDS
  __builtin_amdgcn_global_load_async_to_lds_b128(
      (gptr_b128)(void*)const_cast<bf16*>(gsrc),
      (lptr_b128)(void*)ldst,
      0, 0);
#else
  (void)gsrc; (void)ldst;
#endif
}

__device__ inline v16bf combine16(v8bf lo, v8bf hi) {
  v16bf r;
#pragma unroll
  for (int i = 0; i < 8; ++i) { r[i] = lo[i]; r[i + 8] = hi[i]; }
  return r;
}

__device__ inline v8f zero8() {
  v8f z;
#pragma unroll
  for (int i = 0; i < 8; ++i) z[i] = 0.f;
  return z;
}

// A fragment (16x32 bf16): lane<16: row=r, K={0..7,16..23}; lane>=16: K={8..15,24..31}
__device__ inline v16bf load_frag_a(const bf16* row_k0, int half) {
  return combine16(*(const v8bf*)(row_k0 + half * 8),
                   *(const v8bf*)(row_k0 + 16 + half * 8));
}

// B fragment (32x16 bf16), column-major: lane: col n = lane%16, K = {half*8 + 0..7, 16 + half*8 + 0..7}
__device__ inline v16bf load_frag_b(const bf16* col_k0, int half) {
  return combine16(*(const v8bf*)(col_k0 + half * 8),
                   *(const v8bf*)(col_k0 + 16 + half * 8));
}

__device__ inline v8f wmma_bf16(v16bf a, v16bf b, v8f c) {
  return __builtin_amdgcn_wmma_f32_16x16x32_bf16(false, a, false, b,
                                                 (short)0, c, false, false);
}

// ---------------------------------------------------------------------------
// Elementwise f32 -> bf16
// ---------------------------------------------------------------------------
__global__ __launch_bounds__(256) void cvt_bf16(const float* __restrict__ in,
                                                bf16* __restrict__ out, int n) {
  int i = blockIdx.x * 256 + threadIdx.x;
  if (i < n) out[i] = (__bf16)in[i];
}

// Transposed convert: wt[n*D + k] = (bf16) w[k*D + n]   (D x D)
__global__ __launch_bounds__(256) void cvt_transpose(const float* __restrict__ w,
                                                     bf16* __restrict__ wt) {
  int idx = blockIdx.x * 256 + threadIdx.x;
  int n = idx >> 11;          // D = 2048
  int k = idx & (DM - 1);
  wt[idx] = (__bf16)w[(size_t)k * DM + n];
}

// ---------------------------------------------------------------------------
// Tiled bf16 GEMM:  C[M=4096, N=2048] = A(bf16 row-major) * Bt^T
// Bt is N x K row-major (transposed weights) -> B fragments load column-major.
// Workgroup: 256 thr = 8 waves, 128x128 C tile, k-steps of 64 with a
// double-buffered async (global->LDS) B tile.
// mode 0/1: fused RoPE, store [B,H,T,HD] bf16 (Q / K)
// mode 2  : store transposed  [B,H,HD,T] bf16 (V)
// mode 3  : plain f32 store   [M, N]          (final projection)
// ---------------------------------------------------------------------------
__global__ __launch_bounds__(256) void gemm_bf16(
    const bf16* __restrict__ A, const bf16* __restrict__ Bt,
    const float* __restrict__ cosT, const float* __restrict__ sinT,
    bf16* __restrict__ outb, float* __restrict__ outf, int mode) {
  __shared__ __bf16 ldsB[2 * 128 * BP];

  const int lane = threadIdx.x & 31;
  const int wave = threadIdx.x >> 5;
  const int r    = lane & 15;
  const int half = lane >> 4;
  const int m0   = blockIdx.x * 128 + wave * 16;
  const int n0   = blockIdx.y * 128;

  v8f acc[8];
#pragma unroll
  for (int i = 0; i < 8; ++i) acc[i] = zero8();

  const bf16* Arow = A + (size_t)(m0 + r) * DM;

  // Cooperative B-tile mapping: thread -> (row n, 64B chunk)
  const int bn    = threadIdx.x >> 1;   // 0..127
  const int bpart = threadIdx.x & 1;    // 0..1
  const bf16* gB = Bt + (size_t)(n0 + bn) * DM + bpart * 32;  // +k0 later
  __bf16* lB0 = &ldsB[bn * BP + bpart * 32];
  __bf16* lB1 = lB0 + 128 * BP;

  const int NK = DM / KSTEP;  // 32 stages

#if USE_ASYNC_LDS
  // prologue: prefetch stage 0 into buffer 0
#pragma unroll
  for (int c = 0; c < 4; ++c) async_copy_b128(gB + c * 8, lB0 + c * 8);
#endif

  for (int it = 0; it < NK; ++it) {
    const int k0  = it * KSTEP;
    const int cur = it & 1;
    __bf16* lcur  = cur ? lB1 : lB0;
    __bf16* lnxt  = cur ? lB0 : lB1;

#if USE_ASYNC_LDS
    __builtin_amdgcn_s_wait_asynccnt(0);   // our chunk of buf[cur] landed
    __syncthreads();                        // everyone's chunk landed
    if (it + 1 < NK) {                      // prefetch buf[cur^1]
      const bf16* g = gB + (size_t)(it + 1) * KSTEP;
#pragma unroll
      for (int c = 0; c < 4; ++c) async_copy_b128(g + c * 8, lnxt + c * 8);
    }
    (void)lcur;
#else
    __syncthreads();
    {
      const v8bf* src = (const v8bf*)(gB + k0);
      v8bf t0 = src[0], t1 = src[1], t2 = src[2], t3 = src[3];
      v8bf* dst = (v8bf*)lcur;
      dst[0] = t0; dst[1] = t1; dst[2] = t2; dst[3] = t3;
    }
    __syncthreads();
    (void)lnxt;
#endif

    const __bf16* bufp = cur ? (ldsB + 128 * BP) : ldsB;
#pragma unroll
    for (int kc = 0; kc < 2; ++kc) {
      v16bf af = load_frag_a(Arow + k0 + kc * 32, half);
#pragma unroll
      for (int nt = 0; nt < 8; ++nt) {
        v16bf bfrag = load_frag_b(bufp + (nt * 16 + r) * BP + kc * 32, half);
        acc[nt] = wmma_bf16(af, bfrag, acc[nt]);
      }
    }
  }

  if (mode == 3) {
#pragma unroll
    for (int j = 0; j < 8; ++j) {
      const size_t m = (size_t)(m0 + j + half * 8);
      float* o = outf + m * DM + n0 + r;
#pragma unroll
      for (int nt = 0; nt < 8; ++nt) o[nt * 16] = acc[nt][j];
    }
  } else if (mode == 2) {          // V: store transposed [B,H,HD,T]
    const int h = blockIdx.y;
#pragma unroll
    for (int j = 0; j < 8; ++j) {
      const int m = m0 + j + half * 8;
      const int t = m & (T - 1);
      const int b = m >> 11;
#pragma unroll
      for (int nt = 0; nt < 8; ++nt) {
        const int hd = nt * 16 + r;
        outb[((size_t)(b * H + h) * HD + hd) * T + t] = (__bf16)acc[nt][j];
      }
    }
  } else {                          // Q / K: fused RoPE, [B,H,T,HD]
    const int h = blockIdx.y;
#pragma unroll
    for (int j = 0; j < 8; ++j) {
      const int m = m0 + j + half * 8;
      const int t = m & (T - 1);
      const int b = m >> 11;
#pragma unroll
      for (int nt = 0; nt < 4; ++nt) {
        const int hd = nt * 16 + r;            // 0..63
        const float c  = cosT[t * HALF + hd];
        const float s  = sinT[t * HALF + hd];
        const float x1 = acc[nt][j];
        const float x2 = acc[nt + 4][j];
        acc[nt][j]     = x1 * c - x2 * s;
        acc[nt + 4][j] = x2 * c + x1 * s;
      }
      bf16* o = outb + ((size_t)(b * H + h) * T + t) * HD + r;
#pragma unroll
      for (int nt = 0; nt < 8; ++nt) o[nt * 16] = (__bf16)acc[nt][j];
    }
  }
}

// ---------------------------------------------------------------------------
// Flash attention (causal). One wave per 16-query tile; key blocks of 32.
// Q,K: [B,H,T,HD] bf16.  Vt: [B,H,HD,T] bf16.  Out: [B,T,D] bf16.
// ---------------------------------------------------------------------------
__global__ __launch_bounds__(128) void flash_attn(
    const bf16* __restrict__ Q, const bf16* __restrict__ K,
    const bf16* __restrict__ Vt, bf16* __restrict__ Oout) {
  __shared__ __bf16 ldsP[4 * 16 * LDSP];

  const int lane = threadIdx.x & 31;
  const int wave = threadIdx.x >> 5;
  const int r    = lane & 15;
  const int half = lane >> 4;
  __bf16* Pw = &ldsP[wave * 16 * LDSP];

  const int gw    = blockIdx.x * 4 + wave;   // 0..4095
  const int qtile = gw & 127;                // T/16 tiles
  const int bh    = gw >> 7;                 // 0..31
  const int q0    = qtile * 16;

  const bf16* Qb = Q  + (size_t)bh * T * HD;
  const bf16* Kb = K  + (size_t)bh * T * HD;
  const bf16* Vb = Vt + (size_t)bh * HD * T;

  // resident Q fragments (16 x 128 -> 4 chunks of 16x32)
  const bf16* Qrow = Qb + (size_t)(q0 + r) * HD;
  v16bf qa[4];
#pragma unroll
  for (int kc = 0; kc < 4; ++kc) qa[kc] = load_frag_a(Qrow + kc * 32, half);

  v8f o[8];
#pragma unroll
  for (int i = 0; i < 8; ++i) o[i] = zero8();
  float mrow[8], lrow[8];
#pragma unroll
  for (int j = 0; j < 8; ++j) { mrow[j] = -1e30f; lrow[j] = 0.f; }

  const float scale = 0.08838834764831845f;  // 1/sqrt(128)
  const int kbend = (q0 + 16 + 31) >> 5;     // causal: cover keys 0..q0+15

  for (int kb = 0; kb < kbend; ++kb) {
    const int k0 = kb * 32;
    v8f s0 = zero8(), s1 = zero8();
    const bf16* Kr0 = Kb + (size_t)(k0 + r) * HD;        // keys k0 + n
    const bf16* Kr1 = Kb + (size_t)(k0 + 16 + r) * HD;   // keys k0+16 + n
#pragma unroll
    for (int kc = 0; kc < 4; ++kc) {
      s0 = wmma_bf16(qa[kc], load_frag_b(Kr0 + kc * 32, half), s0);
      s1 = wmma_bf16(qa[kc], load_frag_b(Kr1 + kc * 32, half), s1);
    }

    // online softmax (row lives in a 16-lane half -> butterfly over 16 lanes)
#pragma unroll
    for (int j = 0; j < 8; ++j) {
      const int q = q0 + j + half * 8;
      float v0 = s0[j] * scale;
      float v1 = s1[j] * scale;
      if (k0 + r > q)      v0 = -1e30f;   // causal mask
      if (k0 + 16 + r > q) v1 = -1e30f;
      float mc = fmaxf(v0, v1);
#pragma unroll
      for (int off = 8; off >= 1; off >>= 1)
        mc = fmaxf(mc, __shfl_xor(mc, off, 32));
      const float mn    = fmaxf(mrow[j], mc);
      const float alpha = __expf(mrow[j] - mn);
      const float p0 = __expf(v0 - mn);
      const float p1 = __expf(v1 - mn);
      float rs = p0 + p1;
#pragma unroll
      for (int off = 8; off >= 1; off >>= 1)
        rs += __shfl_xor(rs, off, 32);
      lrow[j] = lrow[j] * alpha + rs;
      mrow[j] = mn;
#pragma unroll
      for (int nt = 0; nt < 8; ++nt) o[nt][j] *= alpha;
      // P (C-layout) -> LDS row-major, for A-fragment reload
      Pw[(j + half * 8) * LDSP + r]      = (__bf16)p0;
      Pw[(j + half * 8) * LDSP + r + 16] = (__bf16)p1;
    }

    // reload P as A fragment (same-wave LDS ops are in-order)
    const __bf16* pp = Pw + r * LDSP + half * 8;
    v16bf pa = combine16(*(const v8bf*)pp, *(const v8bf*)(pp + 16));

    // O += P * V   (V transposed: columns contiguous in key)
#pragma unroll
    for (int nt = 0; nt < 8; ++nt) {
      const bf16* Vcol = Vb + (size_t)(nt * 16 + r) * T + k0;
      o[nt] = wmma_bf16(pa, load_frag_b(Vcol, half), o[nt]);
    }
  }

  // normalize and store merged heads: [B,T,D] bf16
  const int h = bh & (H - 1);
  const int b = bh >> 4;
#pragma unroll
  for (int j = 0; j < 8; ++j) {
    const float inv = 1.f / lrow[j];
    const int t = q0 + j + half * 8;
    bf16* op = Oout + ((size_t)b * T + t) * DM + h * HD + r;
#pragma unroll
    for (int nt = 0; nt < 8; ++nt) op[nt * 16] = (__bf16)(o[nt][j] * inv);
  }
}

// ---------------------------------------------------------------------------
extern "C" void kernel_launch(void* const* d_in, const int* in_sizes, int n_in,
                              void* d_out, int out_size, void* d_ws, size_t ws_size,
                              hipStream_t stream) {
  (void)in_sizes; (void)n_in; (void)out_size; (void)ws_size;
  const float* q_in  = (const float*)d_in[0];
  const float* kv_in = (const float*)d_in[1];
  // d_in[2] = mask (causal handled analytically)
  const float* cosT  = (const float*)d_in[3];
  const float* sinT  = (const float*)d_in[4];
  const float* Wq    = (const float*)d_in[5];
  const float* Wk    = (const float*)d_in[6];
  const float* Wv    = (const float*)d_in[7];
  const float* Wo    = (const float*)d_in[8];
  float* out = (float*)d_out;

  char* ws = (char*)d_ws;
  const size_t MB = 1024ull * 1024ull;
  bf16* Xq   = (bf16*)(ws + 0 * MB);     // [4096,2048] bf16  16MB
  bf16* Xkv  = (bf16*)(ws + 16 * MB);    // 16MB
  bf16* Wqt  = (bf16*)(ws + 32 * MB);    // 8MB each
  bf16* Wkt  = (bf16*)(ws + 40 * MB);
  bf16* Wvt  = (bf16*)(ws + 48 * MB);
  bf16* Wot  = (bf16*)(ws + 56 * MB);
  bf16* Qb   = (bf16*)(ws + 64 * MB);    // [B,H,T,HD]
  bf16* Kb   = (bf16*)(ws + 80 * MB);    // [B,H,T,HD]
  bf16* Vtb  = (bf16*)(ws + 96 * MB);    // [B,H,HD,T]
  bf16* attn = (bf16*)(ws + 112 * MB);   // [B*T, D]

  const int nX = BSZ * T * DM;           // 8388608
  cvt_bf16<<<(nX + 255) / 256, 256, 0, stream>>>(q_in, Xq, nX);
  cvt_bf16<<<(nX + 255) / 256, 256, 0, stream>>>(kv_in, Xkv, nX);
  const int nW = DM * DM;                // 4194304
  cvt_transpose<<<nW / 256, 256, 0, stream>>>(Wq, Wqt);
  cvt_transpose<<<nW / 256, 256, 0, stream>>>(Wk, Wkt);
  cvt_transpose<<<nW / 256, 256, 0, stream>>>(Wv, Wvt);
  cvt_transpose<<<nW / 256, 256, 0, stream>>>(Wo, Wot);

  dim3 gemm_grid(32, 16);                // 4096/128 x 2048/128
  gemm_bf16<<<gemm_grid, 256, 0, stream>>>(Xq,  Wqt, cosT, sinT, Qb,  nullptr, 0);
  gemm_bf16<<<gemm_grid, 256, 0, stream>>>(Xkv, Wkt, cosT, sinT, Kb,  nullptr, 1);
  gemm_bf16<<<gemm_grid, 256, 0, stream>>>(Xkv, Wvt, nullptr, nullptr, Vtb, nullptr, 2);

  flash_attn<<<1024, 128, 0, stream>>>(Qb, Kb, Vtb, attn);

  gemm_bf16<<<gemm_grid, 256, 0, stream>>>(attn, Wot, nullptr, nullptr, nullptr, out, 3);
}
